// FullYoloExport_16990890623226
// MI455X (gfx1250) — matrix operand order repaired and checked by reading, not compiled, not run on hardware
//
#include <hip/hip_runtime.h>
#include <hip/hip_bf16.h>

typedef __attribute__((ext_vector_type(16))) _Float16 v16h;
typedef __attribute__((ext_vector_type(8)))  float    v8f;

#define NCLS   80
#define NANCH  8400
#define KPRE   1000
#define MAXDET 300
#define NBATCH 32

// ---------------------------------------------------------------------------
// Kernel 1: DFL decode via WMMA.
// One wave handles 16 anchors. Two v_wmma_f32_16x16x32_f16 per wave compute
// numerators (Σ exp(x)*bin) and denominators (Σ exp(x)) for all 4 sides.
// ---------------------------------------------------------------------------
__global__ __launch_bounds__(256) void yolo_decode_wmma(
    const float* __restrict__ reg8, const float* __restrict__ reg16,
    const float* __restrict__ reg32, float* __restrict__ boxes)
{
  const int lane = threadIdx.x & 31;
  const int wv   = threadIdx.x >> 5;
  const int g    = blockIdx.x * 8 + wv;     // anchor group of 16
  if (g >= 525) return;                     // 8400/16 groups; wave-uniform exit
  const int b = blockIdx.y;

  const float* reg; int W, HW, strd, rel, baseA;
  if (g < 400)      { reg = reg8;  W = 80; HW = 6400; strd = 8;  rel = g * 16;         baseA = 0;    }
  else if (g < 500) { reg = reg16; W = 40; HW = 1600; strd = 16; rel = (g - 400) * 16; baseA = 6400; }
  else              { reg = reg32; W = 20; HW = 400;  strd = 32; rel = (g - 500) * 16; baseA = 8000; }

  const int  m     = lane & 15;      // anchor within group / output column
  const bool hi    = lane >= 16;
  const int  asRel = rel + m;        // anchor index within this scale
  const int  binBase = hi ? 8 : 0;   // A layout: lo lanes K0-7/16-23, hi lanes K8-15/24-31
  const float* rb = reg + (size_t)b * 64 * HW + asRel;

  // A operands: exp of DFL logits (f16); a1 = sides l,t ; a2 = sides r,b
  v16h a1, a2;
#pragma unroll
  for (int j = 0; j < 8; ++j) {
    a1[j]     = (_Float16)__expf(rb[(size_t)(      binBase + j) * HW]);   // side 0
    a1[j + 8] = (_Float16)__expf(rb[(size_t)(16 +  binBase + j) * HW]);   // side 1
    a2[j]     = (_Float16)__expf(rb[(size_t)(32 +  binBase + j) * HW]);   // side 2
    a2[j + 8] = (_Float16)__expf(rb[(size_t)(48 +  binBase + j) * HW]);   // side 3
  }

  // Constant B: col0 = bin weights K<16, col1 = ones K<16, col2/3 same for K>=16.
  // B layout: lo lanes K0-7/16-23, hi lanes K8-15/24-31 (same as A).
  v16h bm;
#pragma unroll
  for (int j = 0; j < 16; ++j) {
    float bv = 0.f;
    if (j < 8) { bv = (m == 0) ? (float)((j & 7) + binBase) : ((m == 1) ? 1.f : 0.f); }
    else       { bv = (m == 2) ? (float)((j & 7) + binBase) : ((m == 3) ? 1.f : 0.f); }
    bm[j] = (_Float16)bv;
  }

  v8f c0 = {0.f, 0.f, 0.f, 0.f, 0.f, 0.f, 0.f, 0.f};
  v8f c1 = __builtin_amdgcn_wmma_f32_16x16x32_f16(false, a1, false, bm, (short)0, c0, false, false);
  v8f c2 = __builtin_amdgcn_wmma_f32_16x16x32_f16(false, a2, false, bm, (short)0, c0, false, false);

  // C layout: VGPR v <-> row (v + 8*hi), lane <-> column. Stage rows x cols to LDS.
  __shared__ float stage[8][2][128];
  const int rbase = hi ? 8 : 0;
#pragma unroll
  for (int v = 0; v < 8; ++v) {
    stage[wv][0][(v + rbase) * 16 + m] = c1[v];
    stage[wv][1][(v + rbase) * 16 + m] = c2[v];
  }
  __builtin_amdgcn_wave_barrier();   // wave-local LDS exchange; DS ops are in-order

  if (!hi) {
    const float fs = (float)strd;
    float dl = stage[wv][0][m * 16 + 0] / stage[wv][0][m * 16 + 1] * fs;
    float dt = stage[wv][0][m * 16 + 2] / stage[wv][0][m * 16 + 3] * fs;
    float dr = stage[wv][1][m * 16 + 0] / stage[wv][1][m * 16 + 1] * fs;
    float db = stage[wv][1][m * 16 + 2] / stage[wv][1][m * 16 + 3] * fs;
    int h = asRel / W, wq = asRel - h * W;
    float ax = ((float)wq + 0.5f) * fs;
    float ay = ((float)h  + 0.5f) * fs;
    float4 bx = make_float4(ax - dl, ay - dt, ax + dr, ay + db);
    ((float4*)boxes)[(size_t)b * NANCH + baseA + asRel] = bx;
  }
}

// ---------------------------------------------------------------------------
// Kernel 2: exact top-1000 per image via 3-round LDS radix select + bitonic
// sort (key desc, index asc) for deterministic lax.top_k-compatible ordering.
// ---------------------------------------------------------------------------
__device__ __forceinline__ unsigned int mono_key(float f) {
  unsigned int u = __float_as_uint(f);
  return (u & 0x80000000u) ? ~u : (u | 0x80000000u);
}

__global__ __launch_bounds__(1024) void yolo_topk(
    const float* __restrict__ cls8, const float* __restrict__ cls16,
    const float* __restrict__ cls32, unsigned long long* __restrict__ cands)
{
  const int b   = blockIdx.x;
  const int tid = threadIdx.x;
  __shared__ unsigned int hist[4096];
  __shared__ unsigned long long cs[2048];
  __shared__ unsigned int sSel;
  __shared__ int sNeed;
  __shared__ int gCnt, eCnt;

  const float* ptrs[3]  = { cls8  + (size_t)b * NCLS * 6400,
                            cls16 + (size_t)b * NCLS * 1600,
                            cls32 + (size_t)b * NCLS * 400 };
  const int HWs[3]   = { 6400, 1600, 400 };
  const int bases[3] = { 0, 6400, 8000 };
  const int shifts[3] = { 20, 8, 0 };
  const int nbins[3]  = { 4096, 4096, 256 };
  const int dbits[3]  = { 12, 12, 8 };

  unsigned int prefix = 0; int prefBits = 0; int need = KPRE;

  for (int pass = 0; pass < 3; ++pass) {
    const int nb = nbins[pass], sh = shifts[pass];
    for (int i = tid; i < nb; i += 1024) hist[i] = 0;
    __syncthreads();
    for (int s = 0; s < 3; ++s) {
      const int total = NCLS * HWs[s];
      const float* q = ptrs[s];
      for (int p = tid; p < total; p += 1024) {
        unsigned int key = mono_key(q[p]);
        if (prefBits == 0 || (key >> (32 - prefBits)) == prefix)
          atomicAdd(&hist[(key >> sh) & (nb - 1)], 1u);
      }
    }
    __syncthreads();
    if (tid == 0) {
      int cum = 0; unsigned int sel = 0; int nd = need;
      for (int v = nb - 1; v >= 0; --v) {
        int c = (int)hist[v];
        if (cum + c >= need) { sel = (unsigned int)v; nd = need - cum; break; }
        cum += c;
      }
      sSel = sel; sNeed = nd;
    }
    __syncthreads();
    prefix   = (prefix << dbits[pass]) | sSel;
    prefBits += dbits[pass];
    need     = sNeed;
    __syncthreads();
  }
  const unsigned int T = prefix;   // exact key of the 1000th largest; need = rank among equals

  for (int i = tid; i < 2048; i += 1024) cs[i] = 0ull;
  if (tid == 0) { gCnt = 0; eCnt = 0; }
  __syncthreads();

  for (int s = 0; s < 3; ++s) {
    const int HW = HWs[s]; const int total = NCLS * HW; const int base = bases[s];
    const float* q = ptrs[s];
    for (int p = tid; p < total; p += 1024) {
      __builtin_prefetch(q + p + 4096, 0, 1);
      unsigned int key = mono_key(q[p]);
      if (key < T) continue;
      int c = p / HW; int rest = p - c * HW;
      unsigned int idx = (unsigned int)((base + rest) * NCLS + c);
      unsigned long long pk = ((unsigned long long)key << 32) | (unsigned int)(~idx);
      if (key > T) { int sl = atomicAdd(&gCnt, 1); if (sl < 1024) cs[sl] = pk; }
      else         { int sl = atomicAdd(&eCnt, 1); if (sl < 1024) cs[1024 + sl] = pk; }
    }
  }
  __syncthreads();

  // Bitonic sort, descending, 2048 elems (padding sorts last).
  for (int k = 2; k <= 2048; k <<= 1) {
    for (int j = k >> 1; j > 0; j >>= 1) {
      __syncthreads();
      for (int t = tid; t < 2048; t += 1024) {
        int ixj = t ^ j;
        if (ixj > t) {
          unsigned long long x = cs[t], y = cs[ixj];
          bool dsc = ((t & k) == 0);
          if (dsc ? (x < y) : (x > y)) { cs[t] = y; cs[ixj] = x; }
        }
      }
    }
  }
  __syncthreads();
  for (int t = tid; t < KPRE; t += 1024)
    cands[(size_t)b * KPRE + t] = cs[t];
}

// ---------------------------------------------------------------------------
// Kernel 3: per-image greedy NMS (class-offset trick) + stable reorder.
// ---------------------------------------------------------------------------
__global__ __launch_bounds__(1024) void yolo_nms(
    const unsigned long long* __restrict__ cands,
    const float* __restrict__ boxes, float* __restrict__ out)
{
  const int b   = blockIdx.x;
  const int tid = threadIdx.x;
  __shared__ float bx1[KPRE], by1[KPRE], bx2[KPRE], by2[KPRE], sc[KPRE], ar[KPRE];
  __shared__ int   clsArr[KPRE];
  __shared__ unsigned char sup[KPRE], keep[KPRE], vld[KPRE];
  __shared__ int keepFlag;
  __shared__ short order[MAXDET];

  for (int j = tid; j < KPRE; j += 1024) {
    unsigned long long pk = cands[(size_t)b * KPRE + j];
    unsigned int key = (unsigned int)(pk >> 32);
    unsigned int idx = ~((unsigned int)pk);
    unsigned int ub  = (key & 0x80000000u) ? (key & 0x7FFFFFFFu) : ~key;
    float logit = __uint_as_float(ub);
    float s = 1.f / (1.f + __expf(-logit));
    int a = (int)(idx / NCLS); int c = (int)(idx - (unsigned)a * NCLS);
    float4 bx = ((const float4*)boxes)[(size_t)b * NANCH + a];
    bx1[j] = bx.x; by1[j] = bx.y; bx2[j] = bx.z; by2[j] = bx.w;
    sc[j] = s; clsArr[j] = c;
    float w = bx.z - bx.x; w = w > 0.f ? w : 0.f;
    float h = bx.w - bx.y; h = h > 0.f ? h : 0.f;
    ar[j] = w * h;
    sup[j] = 0; keep[j] = 0;
    vld[j] = (s > 0.25f) ? 1 : 0;
  }
  __syncthreads();

  for (int i = 0; i < KPRE; ++i) {
    if (tid == 0) {
      int k = (!sup[i]) && vld[i];
      keep[i] = (unsigned char)k;
      keepFlag = k;
    }
    __syncthreads();
    if (keepFlag) {
      float oi  = (float)clsArr[i] * 4096.0f;
      float xi1 = bx1[i] + oi, yi1 = by1[i] + oi;
      float xi2 = bx2[i] + oi, yi2 = by2[i] + oi;
      float ai  = ar[i];
      for (int j = tid; j < KPRE; j += 1024) {
        float oj  = (float)clsArr[j] * 4096.0f;
        float xx1 = fmaxf(xi1, bx1[j] + oj);
        float yy1 = fmaxf(yi1, by1[j] + oj);
        float xx2 = fminf(xi2, bx2[j] + oj);
        float yy2 = fminf(yi2, by2[j] + oj);
        float iw = xx2 - xx1; iw = iw > 0.f ? iw : 0.f;
        float ih = yy2 - yy1; ih = ih > 0.f ? ih : 0.f;
        float inter = iw * ih;
        float iou = inter / (ai + ar[j] - inter + 1e-7f);
        if (iou > 0.65f) sup[j] = 1;
      }
    }
    __syncthreads();
  }

  if (tid == 0) {  // stable argsort(-final): kept (already desc) then zeros by index
    int s = 0;
    for (int j = 0; j < KPRE && s < MAXDET; ++j) if (keep[j])  order[s++] = (short)j;
    for (int j = 0; j < KPRE && s < MAXDET; ++j) if (!keep[j]) order[s++] = (short)j;
  }
  __syncthreads();

  if (tid < MAXDET) {
    int k = order[tid];
    float* o = out + ((size_t)b * MAXDET + tid) * 6;
    o[0] = bx1[k]; o[1] = by1[k]; o[2] = bx2[k]; o[3] = by2[k];
    o[4] = keep[k] ? sc[k] : 0.f;
    o[5] = (float)clsArr[k];
  }
}

// ---------------------------------------------------------------------------
extern "C" void kernel_launch(void* const* d_in, const int* in_sizes, int n_in,
                              void* d_out, int out_size, void* d_ws, size_t ws_size,
                              hipStream_t stream) {
  (void)in_sizes; (void)n_in; (void)out_size; (void)ws_size;
  const float* cls8  = (const float*)d_in[0];
  const float* reg8  = (const float*)d_in[1];
  const float* cls16 = (const float*)d_in[2];
  const float* reg16 = (const float*)d_in[3];
  const float* cls32 = (const float*)d_in[4];
  const float* reg32 = (const float*)d_in[5];

  float* boxes = (float*)d_ws;                                   // 32*8400*4 f32
  unsigned long long* cands =
      (unsigned long long*)((char*)d_ws + (size_t)NBATCH * NANCH * 4 * sizeof(float));

  yolo_decode_wmma<<<dim3(66, NBATCH), 256, 0, stream>>>(reg8, reg16, reg32, boxes);
  yolo_topk<<<dim3(NBATCH), 1024, 0, stream>>>(cls8, cls16, cls32, cands);
  yolo_nms<<<dim3(NBATCH), 1024, 0, stream>>>(cands, boxes, (float*)d_out);
}
